// GraphUpdateBlock_41446434406631
// MI455X (gfx1250) — compile-verified, matrix operand order checked by reading, not run on hardware
//
#include <hip/hip_runtime.h>

// ---------------------------------------------------------------------------
// GraphUpdateBlock forward for MI455X (gfx1250, wave32).
// GEMMs on the matrix pipe: v_wmma_f32_16x16x32_bf16 (bf16 in, f32 accum),
// B-tile staged in LDS (shared by all 8 waves of the block), A streamed from
// global. Edge scatter-softmax: L2-resident gathers + GLOBAL_ATOMIC_{MAX_U32,ADD_F32}.
// ---------------------------------------------------------------------------

typedef unsigned short ushort_t;
typedef __attribute__((ext_vector_type(16))) __bf16 bf16x16;
typedef __attribute__((ext_vector_type(8)))  __bf16 bf16x8;
typedef __attribute__((ext_vector_type(8)))  float  floatx8;

__device__ __forceinline__ ushort_t f2bf(float f) {           // RNE f32 -> bf16
  unsigned u = __float_as_uint(f);
  u += 0x7FFFu + ((u >> 16) & 1u);
  return (ushort_t)(u >> 16);
}
__device__ __forceinline__ float sigm(float x) { return 1.0f / (1.0f + expf(-x)); }
__device__ __forceinline__ float siluf(float x) { return x / (1.0f + expf(-x)); }

// monotone float<->u32 order-preserving encoding for atomicMax-based segment max
__device__ __forceinline__ unsigned encf(float f) {
  int b = __float_as_int(f);
  return (unsigned)(b ^ ((b >> 31) | 0x80000000));
}
__device__ __forceinline__ float decf(unsigned k) {
  int b = (k & 0x80000000u) ? (int)(k ^ 0x80000000u) : ~(int)k;
  return __int_as_float(b);
}

// block-wide sum for blockDim.x == 256 (8 waves of 32)
__device__ __forceinline__ float blocksum256(float v, float* sb) {
#pragma unroll
  for (int o = 16; o > 0; o >>= 1) v += __shfl_xor(v, o, 32);
  int w = threadIdx.x >> 5, l = threadIdx.x & 31;
  __syncthreads();
  if (l == 0) sb[w] = v;
  __syncthreads();
  float s = 0.f;
#pragma unroll
  for (int j = 0; j < 8; ++j) s += sb[j];
  return s;
}

// ---------------------------------------------------------------------------
// Weight conversion: W[K][Nn] f32 (row major)  ->  Wt[Nn][Kpad] bf16 (row = out ch)
// ---------------------------------------------------------------------------
__global__ void k_convT(const float* __restrict__ W, ushort_t* __restrict__ out,
                        int K, int Kpad, int Nn) {
  int idx = blockIdx.x * 256 + threadIdx.x;
  if (idx >= Nn * Kpad) return;
  int n = idx / Kpad, k = idx - n * Kpad;
  out[idx] = (k < K) ? f2bf(W[(size_t)k * Nn + n]) : (ushort_t)0;
}

// fuse res_W | gat_Wl | gat_Wr ([324][256] each) -> [768][352] bf16 transposed
__global__ void k_fuse3W(const float* __restrict__ W0, const float* __restrict__ W1,
                         const float* __restrict__ W2, ushort_t* __restrict__ out) {
  int idx = blockIdx.x * 256 + threadIdx.x;
  if (idx >= 768 * 352) return;
  int n = idx / 352, k = idx - n * 352;
  const float* W = (n < 256) ? W0 : ((n < 512) ? W1 : W2);
  int col = n & 255;
  out[idx] = (k < 324) ? f2bf(W[(size_t)k * 256 + col]) : (ushort_t)0;
}

__global__ void k_fuse3b(const float* b0, const float* b1, const float* b2, float* out) {
  int t = blockIdx.x * 256 + threadIdx.x;
  if (t >= 768) return;
  out[t] = (t < 256) ? b0[t] : ((t < 512) ? b1[t - 256] : b2[t - 512]);
}

__global__ void k_tobf(const float* __restrict__ in, ushort_t* __restrict__ out, int n) {
  int i = blockIdx.x * 256 + threadIdx.x;
  if (i < n) out[i] = f2bf(in[i]);
}

// ---------------------------------------------------------------------------
// Build x [BN][352] bf16: [node_feat(256) | r/f(2) | tri_w | vp_s | pos_feat(64) | 0-pad(28)]
// One block (256 threads) per node; pos-MLP computed cooperatively.
// ---------------------------------------------------------------------------
__global__ void k_build_x(const float* __restrict__ node_feat, const float* __restrict__ r,
                          const float* __restrict__ tri_w, const float* __restrict__ vp_s,
                          const float* __restrict__ intr, const float* __restrict__ kpts,
                          const float* __restrict__ pts3d,
                          const float* __restrict__ pW1, const float* __restrict__ pb1,
                          const float* __restrict__ pW2, const float* __restrict__ pb2,
                          ushort_t* __restrict__ xbf, int N) {
  int i = blockIdx.x, t = threadIdx.x;
  int b = i / N;
  __shared__ float ph[32];
  __shared__ float uvd[3];
  if (t == 0) {
    uvd[0] = kpts[(size_t)i * 2 + 0] * (1.0f / 1216.0f);
    uvd[1] = kpts[(size_t)i * 2 + 1] * (1.0f / 352.0f);
    uvd[2] = pts3d[(size_t)i * 3 + 2];
  }
  __syncthreads();
  if (t < 32) {
    float s = pb1[t] + uvd[0] * pW1[t] + uvd[1] * pW1[32 + t] + uvd[2] * pW1[64 + t];
    ph[t] = siluf(s);
  }
  __syncthreads();
  ushort_t* row = xbf + (size_t)i * 352;
  row[t] = f2bf(node_feat[(size_t)i * 256 + t]);
  if (t < 64) {
    float s = pb2[t];
#pragma unroll
    for (int j = 0; j < 32; ++j) s += ph[j] * pW2[j * 64 + t];
    row[260 + t] = f2bf(s);
  }
  if (t < 2)  row[256 + t] = f2bf(r[(size_t)i * 2 + t] / intr[b * 4 + t]);
  if (t == 2) row[258] = f2bf(tri_w[i]);
  if (t == 3) row[259] = f2bf(vp_s[i]);
  if (t < 28) row[324 + t] = 0;
}

// ---------------------------------------------------------------------------
// WMMA GEMM: C[M][N] = A[M][K]_bf16 * Bt[N][K]_bf16^T + bias (+optional silu)
// Block = 8 waves = 128 rows x 64 cols. The 64xK B tile (contiguous in Bt)
// is staged once into LDS and shared by all waves (ds_load_b128 fragment
// reads); A streams from global. One wave computes a 16x64 tile with 4
// accumulators (A-frag reused across the 4 N-subtiles).
// A-fragment per ISA 16-bit layout: lane<16 -> K{0..7,16..23}, lane>=16 -> K{8..15,24..31}.
// Requires: M % 128 == 0, N % 64 == 0, K % 32 == 0, K <= 352.
// ---------------------------------------------------------------------------
__global__ void k_gemm(const ushort_t* __restrict__ A, const ushort_t* __restrict__ Bt,
                       const float* __restrict__ bias, float* __restrict__ C,
                       int M, int N, int K, int act) {
  __shared__ __align__(16) ushort_t sB[64 * 352];   // up to 45 KB (of 320 KB WGP LDS)
  int tid = threadIdx.x;
  int wid = tid >> 5, lane = tid & 31;
  int mblocks = M >> 7;
  int mb = blockIdx.x % mblocks;
  int n0 = (blockIdx.x / mblocks) << 6;

  // cooperative stage of Bt[n0 .. n0+63][0..K)  (contiguous 64*K halves)
  {
    const uint4* src = (const uint4*)(Bt + (size_t)n0 * K);
    uint4* dst = (uint4*)sB;
    int chunks = (64 * K) >> 3;          // 16-byte chunks
    for (int c = tid; c < chunks; c += 256) dst[c] = src[c];
  }
  __syncthreads();

  int row = lane & 15;
  int kg = (lane >> 4) << 3;  // 0 or 8
  const ushort_t* arow = A + (size_t)((mb * 8 + wid) * 16 + row) * K;
  floatx8 acc[4] = {floatx8{}, floatx8{}, floatx8{}, floatx8{}};

  for (int k0 = 0; k0 < K; k0 += 32) {
    bf16x8 alo = *(const bf16x8*)(arow + k0 + kg);
    bf16x8 ahi = *(const bf16x8*)(arow + k0 + kg + 16);
    bf16x16 a = __builtin_shufflevector(alo, ahi, 0, 1, 2, 3, 4, 5, 6, 7,
                                        8, 9, 10, 11, 12, 13, 14, 15);
#pragma unroll
    for (int nt = 0; nt < 4; ++nt) {
      const ushort_t* brow = sB + (nt * 16 + row) * K;
      bf16x8 blo = *(const bf16x8*)(brow + k0 + kg);
      bf16x8 bhi = *(const bf16x8*)(brow + k0 + kg + 16);
      bf16x16 bb = __builtin_shufflevector(blo, bhi, 0, 1, 2, 3, 4, 5, 6, 7,
                                           8, 9, 10, 11, 12, 13, 14, 15);
      acc[nt] = __builtin_amdgcn_wmma_f32_16x16x32_bf16(
          false, a, false, bb, (short)0, acc[nt], false, false);
    }
  }
  // D layout: VGPR r holds M=r (lanes 0-15) / M=8+r (lanes 16-31); N = lane&15
  int mhi = (lane >> 4) * 8;
#pragma unroll
  for (int nt = 0; nt < 4; ++nt) {
    int n = n0 + nt * 16 + (lane & 15);
    float bv = bias ? bias[n] : 0.0f;
#pragma unroll
    for (int rI = 0; rI < 8; ++rI) {
      int m = (mb * 8 + wid) * 16 + mhi + rI;
      float v = acc[nt][rI] + bv;
      if (act) v = siluf(v);
      C[(size_t)m * N + n] = v;
    }
  }
}

// ---------------------------------------------------------------------------
// Edge pass A: per-edge logits + segment max (one wave per edge).
// big0 layout: [BN][768] = identity(0:256) | xl(256:512) | xr(512:768)
// lane l: head h = l>>3, channels c = (l&7)*8 .. +7
// ---------------------------------------------------------------------------
__global__ void k_edgeA(const float* __restrict__ big0, const int* __restrict__ edges,
                        const float* __restrict__ eattr, const float* __restrict__ We,
                        const float* __restrict__ att, float* __restrict__ logits,
                        unsigned* __restrict__ segmax, int E) {
  int e = blockIdx.x * 8 + (threadIdx.x >> 5);
  if (e >= E) return;
  int l = threadIdx.x & 31;
  int src = edges[e], dst = edges[E + e];
  int h = l >> 3;
  int ch = h * 64 + ((l & 7) << 3);
  const float* xl = big0 + (size_t)src * 768 + 256 + ch;
  const float* xr = big0 + (size_t)dst * 768 + 512 + ch;
  float ea0 = eattr[(size_t)e * 3], ea1 = eattr[(size_t)e * 3 + 1], ea2 = eattr[(size_t)e * 3 + 2];
  float p = 0.f;
#pragma unroll
  for (int j = 0; j < 8; ++j) {
    float v = xl[j] + xr[j] + ea0 * We[ch + j] + ea1 * We[256 + ch + j] + ea2 * We[512 + ch + j];
    v = (v > 0.f) ? v : 0.2f * v;  // leaky_relu(0.2)
    p += v * att[ch + j];
  }
  for (int o = 1; o < 8; o <<= 1) p += __shfl_xor(p, o, 32);
  if ((l & 7) == 0) {
    logits[(size_t)e * 4 + h] = p;
    atomicMax(segmax + (size_t)dst * 4 + h, encf(p));
  }
}

// Edge pass B: ex = exp(logit - segmax[dst]); denom += ex
__global__ void k_edgeB(const float* __restrict__ logits, const int* __restrict__ edges,
                        const unsigned* __restrict__ segmax, float* __restrict__ exb,
                        float* __restrict__ denom, int E) {
  int idx = blockIdx.x * 256 + threadIdx.x;
  if (idx >= E * 4) return;
  int e = idx >> 2, h = idx & 3;
  int dst = edges[E + e];
  unsigned k = segmax[(size_t)dst * 4 + h];
  float smax = k ? decf(k) : 0.0f;  // empty segment -> 0 (matches isfinite masking)
  float ex = expf(logits[idx] - smax);
  exb[idx] = ex;
  atomicAdd(denom + (size_t)dst * 4 + h, ex);
}

// Edge pass C: aggr[dst] += xl[src] * alpha
__global__ void k_edgeC(const float* __restrict__ big0, const int* __restrict__ edges,
                        const float* __restrict__ exb, const float* __restrict__ denom,
                        float* __restrict__ aggr, int E) {
  int e = blockIdx.x * 8 + (threadIdx.x >> 5);
  if (e >= E) return;
  int l = threadIdx.x & 31;
  int src = edges[e], dst = edges[E + e];
  int h = l >> 3;
  int ch = h * 64 + ((l & 7) << 3);
  float alpha = exb[(size_t)e * 4 + h] / (denom[(size_t)dst * 4 + h] + 1e-16f);
  const float* xl = big0 + (size_t)src * 768 + 256 + ch;
#pragma unroll
  for (int j = 0; j < 8; ++j) atomicAdd(aggr + (size_t)dst * 256 + ch + j, xl[j] * alpha);
}

// LN(gn) + silu + identity -> t1 (bf16);  one block per node
__global__ void k_ln1(const float* __restrict__ aggr, const float* __restrict__ gat_bias,
                      const float* __restrict__ g, const float* __restrict__ bb,
                      const float* __restrict__ big0, ushort_t* __restrict__ t1) {
  int i = blockIdx.x, t = threadIdx.x;
  __shared__ float sb[8];
  float v = aggr[(size_t)i * 256 + t] + gat_bias[t];
  float m = blocksum256(v, sb) * (1.0f / 256.0f);
  float d = v - m;
  float var = blocksum256(d * d, sb) * (1.0f / 256.0f);
  float ln = d * rsqrtf(var + 1e-5f) * g[t] + bb[t];
  float out = siluf(ln) + big0[(size_t)i * 768 + t];  // identity columns 0..255
  t1[(size_t)i * 256 + t] = f2bf(out);
}

// LN(ng) of proj output -> xi (bf16)
__global__ void k_ln2(const float* __restrict__ y2, const float* __restrict__ g,
                      const float* __restrict__ bb, ushort_t* __restrict__ xi) {
  int i = blockIdx.x, t = threadIdx.x;
  __shared__ float sb[8];
  float v = y2[(size_t)i * 256 + t];
  float m = blocksum256(v, sb) * (1.0f / 256.0f);
  float d = v - m;
  float var = blocksum256(d * d, sb) * (1.0f / 256.0f);
  float ln = d * rsqrtf(var + 1e-5f) * g[t] + bb[t];
  xi[(size_t)i * 256 + t] = f2bf(ln);
}

// GRU gate math + LN(nh) -> h_new (f32 output) + bf16 copy for heads
__global__ void k_gru(const float* __restrict__ gi, const float* __restrict__ gh,
                      const float* __restrict__ h0, const float* __restrict__ g,
                      const float* __restrict__ bb, float* __restrict__ hnew,
                      ushort_t* __restrict__ hnew_bf) {
  int i = blockIdx.x, t = threadIdx.x;
  __shared__ float sb[8];
  size_t r0 = (size_t)i * 768;
  float ir = gi[r0 + t], iz = gi[r0 + 256 + t], in_ = gi[r0 + 512 + t];
  float hr = gh[r0 + t], hz = gh[r0 + 256 + t], hn = gh[r0 + 512 + t];
  float rr = sigm(ir + hr);
  float z = sigm(iz + hz);
  float n = tanhf(in_ + rr * hn);
  float pre = (1.0f - z) * n + z * h0[(size_t)i * 256 + t];
  float m = blocksum256(pre, sb) * (1.0f / 256.0f);
  float d = pre - m;
  float var = blocksum256(d * d, sb) * (1.0f / 256.0f);
  float hv = d * rsqrtf(var + 1e-5f) * g[t] + bb[t];
  hnew[(size_t)i * 256 + t] = hv;
  hnew_bf[(size_t)i * 256 + t] = f2bf(hv);
}

// conf head: sigmoid(dot(hw1[i], wh_W2[:,0]) + wh_b2[0]); one wave per node
__global__ void k_conf(const float* __restrict__ hw1, const float* __restrict__ W2,
                       const float* __restrict__ b2, float* __restrict__ out, int BN) {
  int i = blockIdx.x * 8 + (threadIdx.x >> 5);
  if (i >= BN) return;
  int l = threadIdx.x & 31;
  const float* row = hw1 + (size_t)i * 128;
  float s = 0.f;
#pragma unroll
  for (int j = 0; j < 4; ++j) s += row[l + j * 32] * W2[(l + j * 32) * 2];
#pragma unroll
  for (int o = 16; o > 0; o >>= 1) s += __shfl_xor(s, o, 32);
  if (l == 0) out[i] = sigm(s + b2[0]);
}

// a_p head: mean_N(sigmoid(dot)) * 0.1  via atomicAdd per batch
__global__ void k_ap(const float* __restrict__ aph, const float* __restrict__ W2,
                     const float* __restrict__ b2, float* __restrict__ out, int BN, int N) {
  int i = blockIdx.x * 8 + (threadIdx.x >> 5);
  if (i >= BN) return;
  int l = threadIdx.x & 31;
  const float* row = aph + (size_t)i * 64;
  float s = row[l] * W2[l] + row[l + 32] * W2[l + 32];
#pragma unroll
  for (int o = 16; o > 0; o >>= 1) s += __shfl_xor(s, o, 32);
  if (l == 0) atomicAdd(out + (i / N), sigm(s + b2[0]) * (0.1f / (float)N));
}

// a_d head: sigmoid(dot)*0.1 per node
__global__ void k_ad(const float* __restrict__ adh, const float* __restrict__ W2,
                     const float* __restrict__ b2, float* __restrict__ out, int BN) {
  int i = blockIdx.x * 8 + (threadIdx.x >> 5);
  if (i >= BN) return;
  int l = threadIdx.x & 31;
  const float* row = adh + (size_t)i * 64;
  float s = row[l] * W2[l] + row[l + 32] * W2[l + 32];
#pragma unroll
  for (int o = 16; o > 0; o >>= 1) s += __shfl_xor(s, o, 32);
  if (l == 0) out[i] = sigm(s + b2[0]) * 0.1f;
}

// ---------------------------------------------------------------------------
extern "C" void kernel_launch(void* const* d_in, const int* in_sizes, int n_in,
                              void* d_out, int out_size, void* d_ws, size_t ws_size,
                              hipStream_t stream) {
  const int N = 4096, BN = 16384;
  const int E = in_sizes[5] / 2;

  const float* h0   = (const float*)d_in[0];
  const float* nf   = (const float*)d_in[1];
  const float* r    = (const float*)d_in[2];
  const float* triw = (const float*)d_in[3];
  const float* vps  = (const float*)d_in[4];
  const int*   edges = (const int*)d_in[5];
  const float* eattr = (const float*)d_in[6];
  const float* intr  = (const float*)d_in[7];
  const float* kpts  = (const float*)d_in[8];
  const float* pts3d = (const float*)d_in[9];
  // params (dict insertion order)
  const float* pos_W1 = (const float*)d_in[10];
  const float* pos_b1 = (const float*)d_in[11];
  const float* pos_W2 = (const float*)d_in[12];
  const float* pos_b2 = (const float*)d_in[13];
  const float* res_W  = (const float*)d_in[14];
  const float* res_b  = (const float*)d_in[15];
  const float* gat_Wl = (const float*)d_in[16];
  const float* gat_bl = (const float*)d_in[17];
  const float* gat_Wr = (const float*)d_in[18];
  const float* gat_br = (const float*)d_in[19];
  const float* gat_We = (const float*)d_in[20];
  const float* gat_att = (const float*)d_in[21];
  const float* gat_bias = (const float*)d_in[22];
  const float* gn_g = (const float*)d_in[23];
  const float* gn_b = (const float*)d_in[24];
  const float* proj_W = (const float*)d_in[25];
  const float* proj_b = (const float*)d_in[26];
  const float* ng_g = (const float*)d_in[27];
  const float* ng_b = (const float*)d_in[28];
  const float* gru_Wih = (const float*)d_in[29];
  const float* gru_Whh = (const float*)d_in[30];
  const float* gru_bih = (const float*)d_in[31];
  const float* gru_bhh = (const float*)d_in[32];
  const float* nh_g = (const float*)d_in[33];
  const float* nh_b = (const float*)d_in[34];
  const float* wh_W1 = (const float*)d_in[35];
  const float* wh_b1 = (const float*)d_in[36];
  const float* wh_W2 = (const float*)d_in[37];
  const float* wh_b2 = (const float*)d_in[38];
  const float* ap_W1 = (const float*)d_in[39];
  const float* ap_b1 = (const float*)d_in[40];
  const float* ap_W2 = (const float*)d_in[41];
  const float* ap_b2 = (const float*)d_in[42];
  const float* ad_W1 = (const float*)d_in[43];
  const float* ad_b1 = (const float*)d_in[44];
  const float* ad_W2 = (const float*)d_in[45];
  const float* ad_b2 = (const float*)d_in[46];

  // ---- workspace bump allocator (with lifetime-based reuse) ----
  char* ws = (char*)d_ws;
  size_t o = 0;
  auto alloc = [&](size_t b) { size_t rr = o; o += (b + 255) & ~(size_t)255; return rr; };

  size_t o_xbf   = alloc((size_t)BN * 352 * 2);   // x bf16 (dead after big GEMM)
  size_t o_wtf   = alloc((size_t)768 * 352 * 2);  // fused Wt bf16
  size_t o_bfus  = alloc(768 * 4);                // fused bias
  size_t o_projw = alloc((size_t)256 * 256 * 2);
  size_t o_wih   = alloc((size_t)768 * 256 * 2);
  size_t o_whh   = alloc((size_t)768 * 256 * 2);
  size_t o_whw1  = alloc((size_t)128 * 256 * 2);
  size_t o_apw1  = alloc((size_t)64 * 256 * 2);
  size_t o_adw1  = alloc((size_t)64 * 256 * 2);
  size_t o_hfbf  = alloc((size_t)BN * 256 * 2);
  size_t o_big   = alloc((size_t)BN * 768 * 4);   // identity|xl|xr (dead after LN1)
  size_t o_edge  = o;                              // start of edge-region (reused by gh)
  size_t o_log   = alloc((size_t)E * 4 * 4);
  size_t o_ex    = alloc((size_t)E * 4 * 4);
  size_t o_smax  = alloc((size_t)BN * 4 * 4);
  size_t o_den   = alloc((size_t)BN * 4 * 4);
  size_t o_aggr  = alloc((size_t)BN * 256 * 4);
  size_t o_t1    = alloc((size_t)BN * 256 * 2);
  size_t o_y2    = alloc((size_t)BN * 256 * 4);
  size_t o_xibf  = alloc((size_t)BN * 256 * 2);
  // reuses (all prior contents dead by the time these are written):
  size_t o_gi   = o_big;                 // 48 MB, written after big0's last read
  size_t o_gh   = o_edge;                // 48 MB fits inside 48.5 MB edge region
  size_t o_hnew = o_xbf;                 // 8 MB fits x_bf's 11.5 MB
  size_t o_hw1  = o_big;                 // heads reuse gi/gh space after GRU combine
  size_t o_aph  = o_hw1 + (size_t)BN * 128 * 4;
  size_t o_adh  = o_aph + (size_t)BN * 64 * 4;

  float* out_h    = (float*)d_out;                       // [BN,256]
  float* out_conf = out_h + (size_t)BN * 256;            // [BN]
  float* out_ap   = out_conf + BN;                       // [4]
  float* out_ad   = out_ap + 4;                          // [BN]

  // ---- zero accumulators (each launch; harness does not re-poison) ----
  hipMemsetAsync(ws + o_smax, 0, (size_t)BN * 4 * 4, stream);
  hipMemsetAsync(ws + o_den, 0, (size_t)BN * 4 * 4, stream);
  hipMemsetAsync(ws + o_aggr, 0, (size_t)BN * 256 * 4, stream);
  hipMemsetAsync(out_ap, 0, 4 * 4, stream);

  // ---- weight conversion (bf16, transposed, K-padded) ----
  k_fuse3W<<<(768 * 352 + 255) / 256, 256, 0, stream>>>(res_W, gat_Wl, gat_Wr, (ushort_t*)(ws + o_wtf));
  k_fuse3b<<<3, 256, 0, stream>>>(res_b, gat_bl, gat_br, (float*)(ws + o_bfus));
  k_convT<<<(256 * 256 + 255) / 256, 256, 0, stream>>>(proj_W, (ushort_t*)(ws + o_projw), 256, 256, 256);
  k_convT<<<(768 * 256 + 255) / 256, 256, 0, stream>>>(gru_Wih, (ushort_t*)(ws + o_wih), 256, 256, 768);
  k_convT<<<(768 * 256 + 255) / 256, 256, 0, stream>>>(gru_Whh, (ushort_t*)(ws + o_whh), 256, 256, 768);
  k_convT<<<(128 * 256 + 255) / 256, 256, 0, stream>>>(wh_W1, (ushort_t*)(ws + o_whw1), 256, 256, 128);
  k_convT<<<(64 * 256 + 255) / 256, 256, 0, stream>>>(ap_W1, (ushort_t*)(ws + o_apw1), 256, 256, 64);
  k_convT<<<(64 * 256 + 255) / 256, 256, 0, stream>>>(ad_W1, (ushort_t*)(ws + o_adw1), 256, 256, 64);
  k_tobf<<<(BN * 256 + 255) / 256, 256, 0, stream>>>(h0, (ushort_t*)(ws + o_hfbf), BN * 256);

  // ---- build x ----
  k_build_x<<<BN, 256, 0, stream>>>(nf, r, triw, vps, intr, kpts, pts3d,
                                    pos_W1, pos_b1, pos_W2, pos_b2,
                                    (ushort_t*)(ws + o_xbf), N);

  auto gemm = [&](size_t oA, size_t oB, const float* bias, size_t oC, int M, int Nc, int K, int act) {
    int blocks = (M / 128) * (Nc / 64);
    k_gemm<<<blocks, 256, 0, stream>>>((const ushort_t*)(ws + oA),
                                       (const ushort_t*)(ws + oB), bias,
                                       (float*)(ws + oC), M, Nc, K, act);
  };

  // identity | xl | xr in one fused GEMM: [BN,352] x [352,768]
  gemm(o_xbf, o_wtf, (const float*)(ws + o_bfus), o_big, BN, 768, 352, 0);

  // ---- GAT edge phase (L2-resident gathers + atomics) ----
  k_edgeA<<<(E + 7) / 8, 256, 0, stream>>>((const float*)(ws + o_big), edges, eattr,
                                           gat_We, gat_att, (float*)(ws + o_log),
                                           (unsigned*)(ws + o_smax), E);
  k_edgeB<<<(E * 4 + 255) / 256, 256, 0, stream>>>((const float*)(ws + o_log), edges,
                                                   (const unsigned*)(ws + o_smax),
                                                   (float*)(ws + o_ex), (float*)(ws + o_den), E);
  k_edgeC<<<(E + 7) / 8, 256, 0, stream>>>((const float*)(ws + o_big), edges,
                                           (const float*)(ws + o_ex), (const float*)(ws + o_den),
                                           (float*)(ws + o_aggr), E);

  // LN(gn)+silu+identity -> t1 bf16
  k_ln1<<<BN, 256, 0, stream>>>((const float*)(ws + o_aggr), gat_bias, gn_g, gn_b,
                                (const float*)(ws + o_big), (ushort_t*)(ws + o_t1));
  // proj GEMM + LN(ng) -> xi bf16
  gemm(o_t1, o_projw, proj_b, o_y2, BN, 256, 256, 0);
  k_ln2<<<BN, 256, 0, stream>>>((const float*)(ws + o_y2), ng_g, ng_b, (ushort_t*)(ws + o_xibf));

  // ---- GRU gates ----
  gemm(o_xibf, o_wih, gru_bih, o_gi, BN, 768, 256, 0);
  gemm(o_hfbf, o_whh, gru_bhh, o_gh, BN, 768, 256, 0);
  k_gru<<<BN, 256, 0, stream>>>((const float*)(ws + o_gi), (const float*)(ws + o_gh), h0,
                                nh_g, nh_b, out_h, (ushort_t*)(ws + o_hnew));

  // ---- heads (WMMA first layers fused with silu, wave-level second layers) ----
  gemm(o_hnew, o_whw1, wh_b1, o_hw1, BN, 128, 256, 1);
  gemm(o_hnew, o_apw1, ap_b1, o_aph, BN, 64, 256, 1);
  gemm(o_hnew, o_adw1, ad_b1, o_adh, BN, 64, 256, 1);
  k_conf<<<(BN + 7) / 8, 256, 0, stream>>>((const float*)(ws + o_hw1), wh_W2, wh_b2, out_conf, BN);
  k_ap<<<(BN + 7) / 8, 256, 0, stream>>>((const float*)(ws + o_aph), ap_W2, ap_b2, out_ap, BN, N);
  k_ad<<<(BN + 7) / 8, 256, 0, stream>>>((const float*)(ws + o_adh), ad_W2, ad_b2, out_ad, BN);

  (void)n_in; (void)in_sizes; (void)out_size; (void)ws_size;
}
